// TCN_GCN_unit_86148454023265
// MI455X (gfx1250) — compile-verified, matrix operand order checked by reading, not run on hardware
//
#include <hip/hip_runtime.h>
#include <hip/hip_bf16.h>

typedef _Float16 half_t;
typedef __attribute__((ext_vector_type(16))) _Float16 v16h;
typedef __attribute__((ext_vector_type(8)))  _Float16 v8h;
typedef __attribute__((ext_vector_type(8)))  float    v8f;

#define NN 64
#define CC 64
#define TT 256
#define VV 25
#define V1 26
#define SP0 6400   /* T*V  */
#define SP1 6656   /* T*V1 */

__device__ inline v8f wmma16(v16h a, v16h b, v8f c) {
  return __builtin_amdgcn_wmma_f32_16x16x32_f16(false, a, false, b, (short)0, c, false, false);
}

// async global->LDS 16B copy (per-lane addresses), tracked by ASYNCcnt
__device__ inline void async_ld16(unsigned int lds_off, const void* gaddr) {
  asm volatile("global_load_async_to_lds_b128 %0, %1, off" :: "v"(lds_off), "v"(gaddr) : "memory");
}
__device__ inline void async_wait0() {
  asm volatile("s_wait_asynccnt 0" ::: "memory");
}
__device__ inline unsigned int lds_addr_of(const void* p) {
  return (unsigned int)(size_t)p;   // LDS aperture: addr[31:0] is the LDS offset
}

__device__ inline void bn_coef(const float* __restrict__ sums, int c, float cnt,
                               float g, float be, float& sc, float& sh) {
  float m = sums[c*2] / cnt;
  float var = sums[c*2+1] / cnt - m*m;
  sc = g * rsqrtf(var + 1e-5f);
  sh = be - m*sc;
}

// ---------------- K1: xbar[n][c][v] = mean_t x -----------------------------
__global__ __launch_bounds__(256) void k_xbar(const float* __restrict__ x, float* __restrict__ xbar) {
  int idx = blockIdx.x*256 + threadIdx.x;
  if (idx >= NN*CC*VV) return;
  int v = idx % VV; int nc = idx / VV;
  const float* p = x + (size_t)nc*SP0 + v;
  float s = 0.f;
  for (int t = 0; t < TT; ++t) s += p[t*VV];
  xbar[idx] = s * (1.f/TT);
}

// ---------------- K2: adjB f16, B-layout [i][n][o][u32][v32] ---------------
__global__ __launch_bounds__(256) void k_adj(
    const float* __restrict__ xbar, const float* __restrict__ Amat, const float* __restrict__ alpha_p,
    const float* __restrict__ c1w, const float* __restrict__ c1b,
    const float* __restrict__ c2w, const float* __restrict__ c2b,
    const float* __restrict__ c4w, const float* __restrict__ c4b,
    half_t* __restrict__ adjB) {
  __shared__ float x1s[8*VV], x2s[8*VV], ds[8*VV*VV];
  int bi = blockIdx.x; int i3 = bi >> 6; int n = bi & 63;
  int tid = threadIdx.x;
  float alpha = alpha_p[0];
  for (int idx = tid; idx < 8*VV; idx += 256) {
    int rr = idx / VV, vv = idx - rr*VV;
    float s1 = c1b[i3*8 + rr], s2 = c2b[i3*8 + rr];
    for (int cc = 0; cc < CC; ++cc) {
      float xb = xbar[(n*CC + cc)*VV + vv];
      s1 += c1w[(i3*8 + rr)*CC + cc] * xb;
      s2 += c2w[(i3*8 + rr)*CC + cc] * xb;
    }
    x1s[idx] = s1; x2s[idx] = s2;
  }
  __syncthreads();
  for (int idx = tid; idx < 8*VV*VV; idx += 256) {
    int rr = idx / (VV*VV), rem = idx - rr*VV*VV;
    int uu = rem / VV, vv = rem - uu*VV;
    ds[idx] = tanhf(x1s[rr*VV + uu] - x2s[rr*VV + vv]);
  }
  __syncthreads();
  for (int idx = tid; idx < CC*1024; idx += 256) {
    int oo = idx >> 10, rem = idx & 1023;
    int uu = rem >> 5, vv = rem & 31;
    float val = 0.f;
    if (uu < VV && vv < VV) {
      float s = c4b[i3*CC + oo];
      for (int rr = 0; rr < 8; ++rr)
        s += c4w[(i3*CC + oo)*8 + rr] * ds[rr*VV*VV + uu*VV + vv];
      val = s*alpha + Amat[(i3*VV + uu)*VV + vv];
    }
    adjB[((size_t)bi*CC + oo)*1024 + (uu*32 + vv)] = (half_t)val;
  }
}

// ---------------- K3: fused GCN (w3 pconv WMMA -> graph-contraction WMMA) --
__global__ __launch_bounds__(256) void k_gcn(
    const float* __restrict__ x, const float* __restrict__ w3, const float* __restrict__ b3,
    const half_t* __restrict__ adjB, float* __restrict__ y_pre) {
  __shared__ __attribute__((aligned(16))) half_t xls[400*72];     // [p][c]
  __shared__ __attribute__((aligned(16))) half_t w3ls[64*72];     // [o][c]
  __shared__ __attribute__((aligned(16))) half_t x3ls[16*16*40];  // [oo][t][v stride40]
  const int n  = blockIdx.x >> 4;
  const int t0 = (blockIdx.x & 15) * 16;
  const int tid = threadIdx.x, lane = tid & 31, wid = tid >> 5;
  const int r = lane & 15, koff = (lane >> 4) * 8, mhi = (lane >> 4) * 8;

  for (int i = tid; i < 16*16*40; i += 256) x3ls[i] = (half_t)0.f;  // pads stay 0
  for (int i = tid; i < CC*400; i += 256) {
    int c = i / 400, p = i - c*400;
    xls[p*72 + c] = (half_t)x[(((size_t)n*CC + c)*TT + t0)*VV + p];
  }
  __syncthreads();

  for (int ot = 0; ot < 4; ++ot) {
    const int obase = ot*16;
    v8f acc[4];
    #pragma unroll
    for (int q = 0; q < 4; ++q)
      #pragma unroll
      for (int j = 0; j < 8; ++j) acc[q][j] = 0.f;

    for (int i3 = 0; i3 < 3; ++i3) {
      __syncthreads();
      for (int i = tid; i < CC*CC; i += 256) {
        int o = i >> 6, c = i & 63;
        w3ls[o*72 + c] = (half_t)w3[(i3*CC + o)*CC + c];
      }
      __syncthreads();
      // x3 = w3_i @ x + b3 over 25 point-groups of 16 (K=64 -> 2 WMMAs)
      for (int g = wid; g < 25; g += 8) {
        v8f cacc;
        #pragma unroll
        for (int j = 0; j < 8; ++j) cacc[j] = b3[i3*CC + obase + mhi + j];
        int pcol = g*16 + r;
        #pragma unroll
        for (int kb = 0; kb < 64; kb += 32) {
          v16h a, b;
          ((v8h*)&a)[0] = *(const v8h*)(w3ls + (obase + r)*72 + kb + koff);
          ((v8h*)&a)[1] = *(const v8h*)(w3ls + (obase + r)*72 + kb + 16 + koff);
          ((v8h*)&b)[0] = *(const v8h*)(xls + pcol*72 + kb + koff);
          ((v8h*)&b)[1] = *(const v8h*)(xls + pcol*72 + kb + 16 + koff);
          cacc = wmma16(a, b, cacc);
        }
        int tloc = pcol / VV, vv = pcol - tloc*VV;
        #pragma unroll
        for (int j = 0; j < 8; ++j)
          x3ls[(j + mhi)*640 + tloc*40 + vv] = (half_t)cacc[j];
      }
      __syncthreads();
      // z[t][u] += x3[t][v] * adjB[u][v]  (K=32 padded, per channel)
      const half_t* adjp = adjB + (((size_t)i3*NN + n)*CC + obase)*1024;
      #pragma unroll
      for (int q = 0; q < 4; ++q) {
        int item = wid*4 + q;
        int oo = item >> 1, ut = item & 1;
        int ucol = ut*16 + r;
        v16h a, b;
        ((v8h*)&a)[0] = *(const v8h*)(x3ls + oo*640 + r*40 + koff);
        ((v8h*)&a)[1] = *(const v8h*)(x3ls + oo*640 + r*40 + 16 + koff);
        ((v8h*)&b)[0] = *(const v8h*)(adjp + oo*1024 + ucol*32 + koff);
        ((v8h*)&b)[1] = *(const v8h*)(adjp + oo*1024 + ucol*32 + 16 + koff);
        acc[q] = wmma16(a, b, acc[q]);
      }
    }
    #pragma unroll
    for (int q = 0; q < 4; ++q) {
      int item = wid*4 + q;
      int oo = item >> 1, ut = item & 1;
      int u = ut*16 + r;
      if (u < VV) {
        int o = obase + oo;
        #pragma unroll
        for (int j = 0; j < 8; ++j)
          y_pre[(((size_t)n*CC + o)*TT + t0 + j + mhi)*VV + u] = acc[q][j];
      }
    }
  }
}

// ---------------- per-channel sum/sumsq (layout (N,64,S)) ------------------
__global__ __launch_bounds__(256) void k_stats(const float* __restrict__ buf,
                                               float* __restrict__ sums, int S) {
  int c = blockIdx.x;
  __shared__ float rs[256], rs2[256];
  float s = 0.f, s2 = 0.f;
  const float* p = buf + (size_t)c*S;
  size_t nstride = (size_t)CC*S;
  for (int n = 0; n < NN; ++n) {
    const float* pn = p + n*nstride;
    for (int i = threadIdx.x; i < S; i += 256) { float v = pn[i]; s += v; s2 += v*v; }
  }
  rs[threadIdx.x] = s; rs2[threadIdx.x] = s2; __syncthreads();
  for (int off = 128; off; off >>= 1) {
    if (threadIdx.x < off) { rs[threadIdx.x] += rs[threadIdx.x+off]; rs2[threadIdx.x] += rs2[threadIdx.x+off]; }
    __syncthreads();
  }
  if (threadIdx.x == 0) { sums[c*2] = rs[0]; sums[c*2+1] = rs2[0]; }
}

// ---------------- K5: y = relu(bn+x); xm_t[n][p][72] f16 (point-major) -----
__global__ __launch_bounds__(256) void k_xm(const float* __restrict__ y_pre, const float* __restrict__ x,
                                            const float* __restrict__ sums, const float* __restrict__ g,
                                            const float* __restrict__ be, half_t* __restrict__ xm) {
  int idx = blockIdx.x*256 + threadIdx.x;  // (n*64+c)*256+t
  int t = idx & 255;
  int c = (idx >> 8) & 63;
  int n = idx >> 14;
  float sc, sh; bn_coef(sums, c, (float)NN*TT*VV, g[c], be[c], sc, sh);
  const float* yp = y_pre + (size_t)idx*VV;
  const float* xp = x + (size_t)idx*VV;
  half_t* o = xm + ((size_t)n*SP1 + t*V1)*72 + c;   // stride 72 between points
  float accm = 0.f;
  for (int v = 0; v < VV; ++v) {
    float y = fmaxf(yp[v]*sc + sh + xp[v], 0.f);
    accm += y;
    o[(size_t)v*72] = (half_t)y;
  }
  o[(size_t)VV*72] = (half_t)(accm * (1.f/VV));
}

// ---------------- pack 4 branch pconv weights into one 64x64 GEMM ----------
__global__ void k_packW(const float* w0, const float* b0, const float* w1, const float* b1,
                        const float* w2, const float* b2, const float* w3, const float* b3,
                        float* Wcat, float* bcat) {
  int tid = threadIdx.x;
  for (int i = tid; i < 1024; i += 256) {
    Wcat[i] = w0[i]; Wcat[1024+i] = w1[i]; Wcat[2048+i] = w2[i]; Wcat[3072+i] = w3[i];
  }
  if (tid < 16) { bcat[tid] = b0[tid]; bcat[16+tid] = b1[tid]; bcat[32+tid] = b2[tid]; bcat[48+tid] = b3[tid]; }
}

// ------- generic 64x64 pconv: f16 point-major in (async->LDS), f32 out -----
__global__ __launch_bounds__(256) void k_pconv(const half_t* __restrict__ inh /*[n][S][72]*/,
                                               const float* __restrict__ W,
                                               const float* __restrict__ bias,
                                               float* __restrict__ out, int S) {
  __shared__ __attribute__((aligned(16))) half_t ils[256*72];  // 36,864 B, async-filled
  __shared__ __attribute__((aligned(16))) half_t wls[64*72];
  int chunks = S >> 8;
  int n = blockIdx.x / chunks;
  int p0 = (blockIdx.x - n*chunks) * 256;
  int tid = threadIdx.x, lane = tid & 31, wid = tid >> 5;
  int r = lane & 15, koff = (lane >> 4) * 8, mhi = (lane >> 4) * 8;

  // async copy: 256 rows x 144 B, contiguous in global (point-major layout)
  {
    const char* src = (const char*)(inh + ((size_t)n*S + p0)*72);
    unsigned int lb = lds_addr_of(&ils[0]);
    for (int i = tid; i < 2304; i += 256)            // 2304 * 16 B = 36,864 B
      async_ld16(lb + i*16, src + (size_t)i*16);
  }
  for (int i = tid; i < CC*CC; i += 256) {
    int o = i >> 6, c = i & 63;
    wls[o*72 + c] = (half_t)W[i];
  }
  async_wait0();
  __syncthreads();

  int obase = (wid >> 1) * 16;
  for (int q = 0; q < 8; ++q) {
    int g = (wid & 1)*8 + q;
    v8f cacc;
    #pragma unroll
    for (int j = 0; j < 8; ++j) cacc[j] = bias[obase + mhi + j];
    #pragma unroll
    for (int kb = 0; kb < 64; kb += 32) {
      v16h a, b;
      ((v8h*)&a)[0] = *(const v8h*)(wls + (obase + r)*72 + kb + koff);
      ((v8h*)&a)[1] = *(const v8h*)(wls + (obase + r)*72 + kb + 16 + koff);
      ((v8h*)&b)[0] = *(const v8h*)(ils + (g*16 + r)*72 + kb + koff);
      ((v8h*)&b)[1] = *(const v8h*)(ils + (g*16 + r)*72 + kb + 16 + koff);
      cacc = wmma16(a, b, cacc);
    }
    int col = p0 + g*16 + r;
    #pragma unroll
    for (int j = 0; j < 8; ++j)
      out[(size_t)(n*CC + obase + mhi + j)*S + col] = cacc[j];
  }
}

// -- K8: act_t[(n*2+br)][t][v][ci] f16 (ch0..31), maxpool (32..47), copy ----
__global__ __launch_bounds__(256) void k_mid(const float* __restrict__ h_pre, const float* __restrict__ sums,
    const float* __restrict__ g0, const float* __restrict__ be0,
    const float* __restrict__ g1, const float* __restrict__ be1,
    const float* __restrict__ g2, const float* __restrict__ be2,
    half_t* __restrict__ act, float* __restrict__ out_cat) {
  size_t idx = (size_t)blockIdx.x*256 + threadIdx.x;  // over N*64*SP1
  int sp = (int)(idx % SP1); size_t rem = idx / SP1;
  int c = (int)(rem & 63); int n = (int)(rem >> 6);
  float val = h_pre[idx];
  if (c < 48) {
    const float* g  = c < 16 ? g0  : (c < 32 ? g1  : g2);
    const float* be = c < 16 ? be0 : (c < 32 ? be1 : be2);
    float sc, sh; bn_coef(sums, c, (float)NN*SP1, g[c & 15], be[c & 15], sc, sh);
    if (c < 32) {
      int br = (c >> 4) & 1, ci = c & 15;
      act[((size_t)(n*2 + br)*SP1 + sp)*16 + ci] = (half_t)fmaxf(val*sc + sh, 0.f);
    } else {
      int t = sp / V1;
      float m = fmaxf(val*sc + sh, 0.f);
      if (t > 0)      m = fmaxf(m, fmaxf(h_pre[idx - V1]*sc + sh, 0.f));
      if (t < TT - 1) m = fmaxf(m, fmaxf(h_pre[idx + V1]*sc + sh, 0.f));
      out_cat[idx] = m;
    }
  } else {
    out_cat[idx] = val;
  }
}

// ------- K9: temporal conv (k=5, pad 2) via WMMA, K=80->96, async im2col ---
__global__ __launch_bounds__(256) void k_tconv(const half_t* __restrict__ act /*[(n*2+br)][t][v][16]*/,
    const float* __restrict__ wt0, const float* __restrict__ bt0,
    const float* __restrict__ wt1, const float* __restrict__ bt1,
    float* __restrict__ out_cat) {
  __shared__ __attribute__((aligned(16))) half_t als[16*96];   // A: [o][kappa=dk*16+ci], pad 80..95 = 0
  __shared__ __attribute__((aligned(16))) half_t els[21*416];  // [t'][v][ci], 17,472 B
  int b = blockIdx.x;
  int t0 = (b & 15) * 16; b >>= 4;
  int br = b & 1; int n = b >> 1;
  const float* wt = br ? wt1 : wt0;
  const float* bt = br ? bt1 : bt0;
  int tid = threadIdx.x, lane = tid & 31, wid = tid >> 5;
  int r = lane & 15, koff = (lane >> 4) * 8, mhi = (lane >> 4) * 8;

  for (int i = tid; i < 16*96; i += 256) {
    int o = i / 96, kk = i - o*96;
    int dk = kk >> 4, ci = kk & 15;
    als[i] = (half_t)((dk < 5) ? wt[(o*16 + ci)*5 + dk] : 0.f);
  }
  // rows t' = t0-2 .. t0+18 (21 rows of 832 B); valid range async-copied, rest zeroed
  int r_lo = (t0 == 0) ? 2 : 0;
  int r_hi = (258 - t0 < 21) ? (258 - t0) : 21;
  for (int i = tid; i < 21*416; i += 256) {
    int row = i / 416;
    if (row < r_lo || row >= r_hi) els[i] = (half_t)0.f;
  }
  {
    const char* src = (const char*)(act + ((size_t)(n*2 + br)*TT + (t0 - 2 + r_lo))*416);
    unsigned int lb = lds_addr_of(&els[0]) + r_lo*832;
    int nbytes = (r_hi - r_lo)*832;                 // multiple of 16
    for (int off = tid*16; off < nbytes; off += 256*16)
      async_ld16(lb + off, src + off);
  }
  async_wait0();
  __syncthreads();

  for (int g = wid; g < 26; g += 8) {
    v8f cacc;
    #pragma unroll
    for (int j = 0; j < 8; ++j) cacc[j] = bt[mhi + j];
    int pcol = g*16 + r;
    int tloc = pcol / V1, vv = pcol - tloc*V1;
    #pragma unroll
    for (int kb = 0; kb < 96; kb += 32) {
      v16h a, bb;
      ((v8h*)&a)[0] = *(const v8h*)(als + r*96 + kb + koff);
      ((v8h*)&a)[1] = *(const v8h*)(als + r*96 + kb + 16 + koff);
      int k1 = kb + koff, k2 = kb + koff + 16;
      ((v8h*)&bb)[0] = *(const v8h*)(els + ((tloc + (k1 >> 4))*V1 + vv)*16 + (k1 & 15));
      ((v8h*)&bb)[1] = *(const v8h*)(els + ((tloc + (k2 >> 4))*V1 + vv)*16 + (k2 & 15));
      cacc = wmma16(a, bb, cacc);
    }
    #pragma unroll
    for (int j = 0; j < 8; ++j)
      out_cat[(((size_t)n*CC + br*16 + mhi + j)*TT + t0 + tloc)*V1 + vv] = cacc[j];
  }
}

// ---------------- K11: feat = bn?(local) + bn?(glob)*coeff -----------------
__global__ __launch_bounds__(256) void k_feat(const float* __restrict__ out_cat, const float* __restrict__ sums,
    const float* __restrict__ gt0, const float* __restrict__ bet0,
    const float* __restrict__ gt1, const float* __restrict__ bet1,
    const float* __restrict__ coeff, float* __restrict__ feat) {
  int idx = blockIdx.x*256 + threadIdx.x;  // (n*64+c)*256+t
  int c = (idx >> 8) & 63;
  float sc = 1.f, sh = 0.f;
  if (c < 32) {
    const float* g  = c < 16 ? gt0  : gt1;
    const float* be = c < 16 ? bet0 : bet1;
    bn_coef(sums, c, (float)NN*SP1, g[c & 15], be[c & 15], sc, sh);
  }
  const float* p = out_cat + (size_t)idx*V1;
  float glob = p[VV]*sc + sh;
  float* o = feat + (size_t)idx*VV;
  for (int v = 0; v < VV; ++v) o[v] = p[v]*sc + sh + glob*coeff[v];
}

// ---------------- K13: g16_t[n][sp][72] = f16(relu(bn(feat))) --------------
__global__ __launch_bounds__(256) void k_g(const float* __restrict__ feat, const float* __restrict__ sums,
                                           const float* __restrict__ tg, const float* __restrict__ tb,
                                           half_t* __restrict__ g16) {
  size_t idx = (size_t)blockIdx.x*256 + threadIdx.x;
  int sp = (int)(idx % SP0);
  int c  = (int)((idx / SP0) & 63);
  int n  = (int)(idx / ((size_t)SP0*64));
  float sc, sh; bn_coef(sums, c, (float)NN*SP0, tg[c], tb[c], sc, sh);
  g16[((size_t)n*SP0 + sp)*72 + c] = (half_t)fmaxf(feat[idx]*sc + sh, 0.f);
}

// ---------------- K16: out = relu(bn(feat2) + x) ---------------------------
__global__ __launch_bounds__(256) void k_out(const float* __restrict__ feat2, const float* __restrict__ x,
                                             const float* __restrict__ sums, const float* __restrict__ g,
                                             const float* __restrict__ be, float* __restrict__ out) {
  size_t idx = (size_t)blockIdx.x*256 + threadIdx.x;
  int c = (int)((idx / SP0) & 63);
  float sc, sh; bn_coef(sums, c, (float)NN*SP0, g[c], be[c], sc, sh);
  out[idx] = fmaxf(feat2[idx]*sc + sh + x[idx], 0.f);
}

extern "C" void kernel_launch(void* const* d_in, const int* in_sizes, int n_in,
                              void* d_out, int out_size, void* d_ws, size_t ws_size,
                              hipStream_t stream) {
  if (n_in < 42) return;
  const float* x      = (const float*)d_in[0];
  const float* Amat   = (const float*)d_in[1];
  const float* alpha  = (const float*)d_in[2];
  const float* c1w    = (const float*)d_in[3];
  const float* c1b    = (const float*)d_in[4];
  const float* c2w    = (const float*)d_in[5];
  const float* c2b    = (const float*)d_in[6];
  const float* c3w    = (const float*)d_in[7];
  const float* c3b    = (const float*)d_in[8];
  const float* c4w    = (const float*)d_in[9];
  const float* c4b    = (const float*)d_in[10];
  const float* gcn_g  = (const float*)d_in[11];
  const float* gcn_b  = (const float*)d_in[12];
  const float* coeff  = (const float*)d_in[13];
  const float* br0w1  = (const float*)d_in[14];
  const float* br0b1  = (const float*)d_in[15];
  const float* br0g1  = (const float*)d_in[16];
  const float* br0be1 = (const float*)d_in[17];
  const float* br0wt  = (const float*)d_in[18];
  const float* br0bt  = (const float*)d_in[19];
  const float* br0gt  = (const float*)d_in[20];
  const float* br0bet = (const float*)d_in[21];
  const float* br1w1  = (const float*)d_in[22];
  const float* br1b1  = (const float*)d_in[23];
  const float* br1g1  = (const float*)d_in[24];
  const float* br1be1 = (const float*)d_in[25];
  const float* br1wt  = (const float*)d_in[26];
  const float* br1bt  = (const float*)d_in[27];
  const float* br1gt  = (const float*)d_in[28];
  const float* br1bet = (const float*)d_in[29];
  const float* br2w1  = (const float*)d_in[30];
  const float* br2b1  = (const float*)d_in[31];
  const float* br2g1  = (const float*)d_in[32];
  const float* br2be1 = (const float*)d_in[33];
  const float* br3w1  = (const float*)d_in[34];
  const float* br3b1  = (const float*)d_in[35];
  const float* t_g    = (const float*)d_in[36];
  const float* t_b    = (const float*)d_in[37];
  const float* t_w    = (const float*)d_in[38];
  const float* t_bias = (const float*)d_in[39];
  const float* tcn_g  = (const float*)d_in[40];
  const float* tcn_b  = (const float*)d_in[41];

  char* ws = (char*)d_ws;
  // region A: y_pre -> h_pre -> feat -> feat2 (disjoint lifetimes)
  const size_t szA = 109051904;
  // region B: adjB (25.2MB) -> xm_t (61.3MB) -> out_cat (109MB)
  const size_t szB = 109051904;
  // region C: xbar (0.4MB) -> act_t (27.3MB) -> g16_t (59.0MB)
  const size_t szC = 58982400;
  float*  y_pre   = (float*)(ws);
  float*  h_pre   = (float*)(ws);
  float*  feat    = (float*)(ws);
  float*  feat2   = (float*)(ws);
  half_t* adjB    = (half_t*)(ws + szA);
  half_t* xm      = (half_t*)(ws + szA);
  float*  out_cat = (float*)(ws + szA);
  float*  xbar    = (float*)(ws + szA + szB);
  half_t* act     = (half_t*)(ws + szA + szB);
  half_t* g16     = (half_t*)(ws + szA + szB);
  float*  Wcat    = (float*)(ws + szA + szB + szC);
  float*  bcat    = (float*)(ws + szA + szB + szC + 16384);
  float*  sums    = (float*)(ws + szA + szB + szC + 16384 + 1024);
  (void)ws_size; (void)in_sizes; (void)out_size;

  // ---- GCN stage ----
  k_xbar <<<(NN*CC*VV + 255)/256, 256, 0, stream>>>(x, xbar);
  k_adj  <<<3*NN, 256, 0, stream>>>(xbar, Amat, alpha, c1w, c1b, c2w, c2b, c4w, c4b, adjB);
  k_gcn  <<<NN*16, 256, 0, stream>>>(x, c3w, c3b, adjB, y_pre);
  k_stats<<<64, 256, 0, stream>>>(y_pre, sums, SP0);
  k_xm   <<<NN*CC*TT/256, 256, 0, stream>>>(y_pre, x, sums, gcn_g, gcn_b, xm);

  // ---- multi-branch TCN entry (one fused 64x64 GEMM) ----
  k_packW<<<1, 256, 0, stream>>>(br0w1, br0b1, br1w1, br1b1, br2w1, br2b1, br3w1, br3b1, Wcat, bcat);
  k_pconv<<<NN*(SP1/256), 256, 0, stream>>>(xm, Wcat, bcat, h_pre, SP1);
  k_stats<<<48, 256, 0, stream>>>(h_pre, sums, SP1);
  k_mid  <<<(int)(((size_t)NN*CC*SP1)/256), 256, 0, stream>>>(h_pre, sums, br0g1, br0be1, br1g1, br1be1,
                                                              br2g1, br2be1, act, out_cat);
  k_tconv<<<NN*2*16, 256, 0, stream>>>(act, br0wt, br0bt, br1wt, br1bt, out_cat);
  k_stats<<<32, 256, 0, stream>>>(out_cat, sums, SP1);

  // ---- combine + final pconv ----
  k_feat <<<NN*CC*TT/256, 256, 0, stream>>>(out_cat, sums, br0gt, br0bet, br1gt, br1bet, coeff, feat);
  k_stats<<<64, 256, 0, stream>>>(feat, sums, SP0);
  k_g    <<<(int)(((size_t)NN*CC*SP0)/256), 256, 0, stream>>>(feat, sums, t_g, t_b, g16);
  k_pconv<<<NN*(SP0/256), 256, 0, stream>>>(g16, t_w, t_bias, feat2, SP0);
  k_stats<<<64, 256, 0, stream>>>(feat2, sums, SP0);
  k_out  <<<(int)(((size_t)NN*CC*SP0)/256), 256, 0, stream>>>(feat2, x, sums, tcn_g, tcn_b, (float*)d_out);
}
